// SlidingWindowAttention_75033078661337
// MI455X (gfx1250) — compile-verified, matrix operand order checked by reading, not run on hardware
//
#include <hip/hip_runtime.h>
#include <hip/hip_bf16.h>
#include <stdint.h>

#define BATCH 2
#define S_LEN 4096
#define DIM   1024
#define HEADS 16
#define HD    64
#define WIN   512
#define NBLK  8

typedef __bf16 bf16;
typedef __attribute__((ext_vector_type(16))) __bf16 v16bf;
typedef __attribute__((ext_vector_type(8)))  float  v8f;
typedef __attribute__((ext_vector_type(4)))  int    v4i;

// Async global->LDS path (CDNA5). Guarded so the file still compiles if the
// builtins are not declared by this toolchain; fallback is synchronous b128.
#if defined(__HIP_DEVICE_COMPILE__) && defined(__gfx1250__) && \
    __has_builtin(__builtin_amdgcn_global_load_async_to_lds_b128) && \
    __has_builtin(__builtin_amdgcn_s_wait_asynccnt)
#define USE_ASYNC_LDS 1
#else
#define USE_ASYNC_LDS 0
#endif

// ISA 7.12.2: 16-bit A/B fragment K index for vgpr-pair v and lane-half.
__device__ __forceinline__ int kk_of(int v, int half) {
  return ((v >> 2) << 4) + (half << 3) + ((v & 3) << 1);
}

__device__ __forceinline__ v8f v8f_zero() {
  v8f z = {0.f, 0.f, 0.f, 0.f, 0.f, 0.f, 0.f, 0.f};
  return z;
}

__device__ __forceinline__ void copy16_to_lds(const bf16* g, bf16* l) {
#if USE_ASYNC_LDS
  __builtin_amdgcn_global_load_async_to_lds_b128(
      (__attribute__((address_space(1))) v4i*)g,
      (__attribute__((address_space(3))) v4i*)l, 0, 0);
#else
  *(uint4*)l = *(const uint4*)g;
#endif
}

__device__ __forceinline__ void wait_stage(bool more_inflight) {
#if USE_ASYNC_LDS
  if (more_inflight) __builtin_amdgcn_s_wait_asynccnt(4);
  else               __builtin_amdgcn_s_wait_asynccnt(0);
#else
  (void)more_inflight;
#endif
}

// ---------------------------------------------------------------------------
// fp32 -> bf16 convert
// ---------------------------------------------------------------------------
__global__ void swa_cvt_bf16(const float* __restrict__ in, bf16* __restrict__ out, int n) {
  int i = blockIdx.x * blockDim.x + threadIdx.x;
  int stride = gridDim.x * blockDim.x;
  for (; i < n; i += stride) out[i] = (bf16)in[i];
}

// ---------------------------------------------------------------------------
// Head-wise V transpose: V[b*S+s][h*64+e] -> Vt[(b*16+h)*64+e][s]
// 32x32 LDS tiles so both the read and the write are contiguous.
// ---------------------------------------------------------------------------
__global__ __launch_bounds__(256) void swa_transpose_v(
    const bf16* __restrict__ V, bf16* __restrict__ Vt) {
  __shared__ bf16 tile[32][33];
  const int s0 = blockIdx.x * 32;            // 0..4095 step 32
  const int e0 = blockIdx.y * 32;            // 0 or 32
  const int bh = blockIdx.z;                 // 0..31
  const int b = bh >> 4, h = bh & 15;
  const int c = threadIdx.x & 31, q = threadIdx.x >> 5;   // q: 0..7

#pragma unroll
  for (int i = 0; i < 4; ++i) {
    const int r = q + i * 8;
    tile[r][c] = V[((size_t)b * S_LEN + s0 + r) * DIM + h * HD + e0 + c];
  }
  __syncthreads();
#pragma unroll
  for (int i = 0; i < 4; ++i) {
    const int a = q + i * 8;
    Vt[((size_t)bh * HD + e0 + a) * S_LEN + s0 + c] = tile[c][a];
  }
}

// ---------------------------------------------------------------------------
// Y[M,1024] = A[M,1024](bf16) @ W[1024,1024]^T (bf16) + bias(f32)
// 128x128 workgroup tile, 8 waves of 32x64, K-step 32, double-buffered LDS
// with async global->LDS staging.
// ---------------------------------------------------------------------------
template <int OUT_F32>
__global__ __launch_bounds__(256) void swa_gemm_bias(
    const bf16* __restrict__ A, const bf16* __restrict__ Bw,
    const float* __restrict__ bias, void* __restrict__ Out) {
  __shared__ __align__(16) bf16 As[2][128 * 32];
  __shared__ __align__(16) bf16 Bs[2][128 * 32];

  const int tid  = threadIdx.x;
  const int wid  = tid >> 5, lane = tid & 31;
  const int half = lane >> 4, ln = lane & 15;
  const int wm   = wid >> 1, wn = wid & 1;
  const int m0   = blockIdx.y * 128, n0 = blockIdx.x * 128;

  v8f acc[2][4];
#pragma unroll
  for (int mt = 0; mt < 2; ++mt)
#pragma unroll
    for (int nt = 0; nt < 4; ++nt) acc[mt][nt] = v8f_zero();

  auto stage = [&](int buf, int k0) {
    for (int i = tid; i < 512; i += 256) {
      const int r = i >> 2, seg = i & 3;
      copy16_to_lds(A  + (size_t)(m0 + r) * DIM + k0 + seg * 8, &As[buf][r * 32 + seg * 8]);
      copy16_to_lds(Bw + (size_t)(n0 + r) * DIM + k0 + seg * 8, &Bs[buf][r * 32 + seg * 8]);
    }
  };

  stage(0, 0);
  for (int kt = 0; kt < DIM / 32; ++kt) {
    const int  buf  = kt & 1;
    const bool more = (kt + 1) < (DIM / 32);
    if (more) stage(buf ^ 1, (kt + 1) * 32);
    wait_stage(more);
    __syncthreads();

    const uint32_t* Au = (const uint32_t*)As[buf];
    const uint32_t* Bu = (const uint32_t*)Bs[buf];
    v16bf afrag[2];
#pragma unroll
    for (int mt = 0; mt < 2; ++mt) {
      uint32_t* au = (uint32_t*)&afrag[mt];
      const int row = wm * 32 + mt * 16 + ln;
#pragma unroll
      for (int v = 0; v < 8; ++v) au[v] = Au[row * 16 + (kk_of(v, half) >> 1)];
    }
#pragma unroll
    for (int nt = 0; nt < 4; ++nt) {
      v16bf bfrag;
      uint32_t* bu = (uint32_t*)&bfrag;
      const int col = wn * 64 + nt * 16 + ln;
#pragma unroll
      for (int v = 0; v < 8; ++v) bu[v] = Bu[col * 16 + (kk_of(v, half) >> 1)];
#pragma unroll
      for (int mt = 0; mt < 2; ++mt)
        acc[mt][nt] = __builtin_amdgcn_wmma_f32_16x16x32_bf16(
            false, afrag[mt], false, bfrag, (short)0, acc[mt][nt], false, false);
    }
    __syncthreads();
  }

#pragma unroll
  for (int nt = 0; nt < 4; ++nt) {
    const int   col = n0 + wn * 64 + nt * 16 + ln;
    const float bv  = bias[col];
#pragma unroll
    for (int mt = 0; mt < 2; ++mt) {
#pragma unroll
      for (int v = 0; v < 8; ++v) {
        const int   row = m0 + wm * 32 + mt * 16 + v + 8 * half;
        const float val = acc[mt][nt][v] + bv;
        if (OUT_F32) ((float*)Out)[(size_t)row * DIM + col] = val;
        else         ((bf16*)Out)[(size_t)row * DIM + col]  = (bf16)val;
      }
    }
  }
}

// ---------------------------------------------------------------------------
// Sliding-window attention core. One workgroup = (b, h, window block n,
// 16-query tile). 8 waves: scores for 128 keys each, then P@V with K split.
// V is consumed pre-transposed (Vt[bh][e][s]) so PV fragments are b128 loads.
// ---------------------------------------------------------------------------
__global__ __launch_bounds__(256) void swa_attn(
    const bf16* __restrict__ Q, const bf16* __restrict__ Kt,
    const bf16* __restrict__ Vt, bf16* __restrict__ AO) {
  __shared__ __align__(16) bf16 Pl[16 * 1024];   // exp(P) bf16, 32KB
  __shared__ float red[8][16];
  __shared__ float rowmax_s[16];
  __shared__ float rowsum_s[16];
  __shared__ float outred[16 * 64];

  const int tid  = threadIdx.x;
  const int wid  = tid >> 5, lane = tid & 31;
  const int half = lane >> 4, ln = lane & 15;
  const int qsub = blockIdx.x;                 // 0..31 (16 queries each)
  const int nblk = blockIdx.y;                 // 0..7
  const int b    = blockIdx.z >> 4, h = blockIdx.z & 15;
  const int colh = h * HD;
  const size_t rowQ0 = (size_t)b * S_LEN + (size_t)nblk * WIN + qsub * 16;
  const int kwin0 = nblk * WIN - WIN;          // abs key pos at window index 0

  for (int i = tid; i < 16 * 64; i += 256) outred[i] = 0.f;

  // ---- scores: S = Q(16xhd) @ Kwin(2W x hd)^T, this wave: keys [wid*128, +128)
  v8f acc[8];
#pragma unroll
  for (int nt = 0; nt < 8; ++nt) acc[nt] = v8f_zero();

#pragma unroll
  for (int ks = 0; ks < 2; ++ks) {
    v16bf a;
    uint32_t* au = (uint32_t*)&a;
#pragma unroll
    for (int v = 0; v < 8; ++v)
      au[v] = *(const uint32_t*)(Q + (rowQ0 + ln) * DIM + colh + ks * 32 + kk_of(v, half));
#pragma unroll
    for (int nt = 0; nt < 8; ++nt) {
      const int t = wid * 128 + nt * 16 + ln;
      int absk = kwin0 + t; if (absk < 0) absk = 0;      // masked below anyway
      const bf16* kp = Kt + ((size_t)b * S_LEN + absk) * DIM + colh + ks * 32;
      __builtin_prefetch(kp + DIM, 0, 1);                // global_prefetch_b8
      v16bf bb;
      uint32_t* bu = (uint32_t*)&bb;
#pragma unroll
      for (int v = 0; v < 8; ++v)
        bu[v] = *(const uint32_t*)(kp + kk_of(v, half));
      acc[nt] = __builtin_amdgcn_wmma_f32_16x16x32_bf16(
          false, a, false, bb, (short)0, acc[nt], false, false);
    }
  }

  // ---- mask + scale + row max ----
  const float scale = 0.125f;   // hd^-0.5
  float rmax[8];
#pragma unroll
  for (int v = 0; v < 8; ++v) rmax[v] = -3.0e38f;
#pragma unroll
  for (int nt = 0; nt < 8; ++nt) {
    const int t = wid * 128 + nt * 16 + ln;
#pragma unroll
    for (int v = 0; v < 8; ++v) {
      const int s_ = qsub * 16 + v + 8 * half;           // query pos in W block
      const bool valid = (t <= s_ + WIN) && (t >= s_ + 1) && (kwin0 + t >= 0);
      const float sc = valid ? acc[nt][v] * scale : -1.0e9f;
      acc[nt][v] = sc;
      rmax[v] = fmaxf(rmax[v], sc);
    }
  }
#pragma unroll
  for (int m = 1; m < 16; m <<= 1)
#pragma unroll
    for (int v = 0; v < 8; ++v) rmax[v] = fmaxf(rmax[v], __shfl_xor(rmax[v], m, 32));
  if (ln == 0)
#pragma unroll
    for (int v = 0; v < 8; ++v) red[wid][v + 8 * half] = rmax[v];
  __syncthreads();
  if (tid < 16) {
    float m = red[0][tid];
    for (int w = 1; w < 8; ++w) m = fmaxf(m, red[w][tid]);
    rowmax_s[tid] = m;
  }
  __syncthreads();

  // ---- exp, row sum, stash P (bf16) in LDS ----
  float rmaxl[8], rsum[8];
#pragma unroll
  for (int v = 0; v < 8; ++v) { rmaxl[v] = rowmax_s[v + 8 * half]; rsum[v] = 0.f; }
#pragma unroll
  for (int nt = 0; nt < 8; ++nt) {
#pragma unroll
    for (int v = 0; v < 8; ++v) {
      const float p = __expf(acc[nt][v] - rmaxl[v]);
      rsum[v] += p;
      Pl[(v + 8 * half) * 1024 + wid * 128 + nt * 16 + ln] = (bf16)p;
    }
  }
#pragma unroll
  for (int m = 1; m < 16; m <<= 1)
#pragma unroll
    for (int v = 0; v < 8; ++v) rsum[v] += __shfl_xor(rsum[v], m, 32);
  if (ln == 0)
#pragma unroll
    for (int v = 0; v < 8; ++v) red[wid][v + 8 * half] = rsum[v];
  __syncthreads();
  if (tid < 16) {
    float s = red[0][tid];
    for (int w = 1; w < 8; ++w) s += red[w][tid];
    rowsum_s[tid] = s;
  }
  __syncthreads();

  // ---- out = P(16x1024) @ Vwin(1024x64); wave: one 16x16 tile, half of K ----
  const int khalf = wid >> 2, ntile = wid & 3;
  const int e = ntile * 16 + ln;
  const bf16* vtb = Vt + ((size_t)blockIdx.z * HD + e) * S_LEN;  // row e of Vt
  v8f oacc = v8f_zero();
  const uint32_t* Pu = (const uint32_t*)Pl;
  for (int ks = 0; ks < 16; ++ks) {
    const int kb = khalf * 512 + ks * 32;
    v16bf a;
    uint32_t* au = (uint32_t*)&a;
#pragma unroll
    for (int v = 0; v < 8; ++v)
      au[v] = Pu[ln * 512 + ((kb + kk_of(v, half)) >> 1)];
    v16bf bb;
    uint32_t* bu = (uint32_t*)&bb;
#pragma unroll
    for (int v = 0; v < 8; ++v) {
      int k0 = kwin0 + kb + kk_of(v, half);   // even; pair never straddles 0
      if (k0 < 0) k0 = 0;                     // P==0 there, product is 0
      bu[v] = *(const uint32_t*)(vtb + k0);
    }
    oacc = __builtin_amdgcn_wmma_f32_16x16x32_bf16(
        false, a, false, bb, (short)0, oacc, false, false);
  }
#pragma unroll
  for (int v = 0; v < 8; ++v)
    atomicAdd(&outred[(v + 8 * half) * 64 + e], oacc[v]);
  __syncthreads();

  for (int i = tid; i < 16 * 64; i += 256) {
    const int r = i >> 6, c = i & 63;
    AO[(rowQ0 + r) * DIM + colh + c] = (bf16)(outred[i] / rowsum_s[r]);
  }
}

// ---------------------------------------------------------------------------
// Launcher
// ---------------------------------------------------------------------------
extern "C" void kernel_launch(void* const* d_in, const int* in_sizes, int n_in,
                              void* d_out, int out_size, void* d_ws, size_t ws_size,
                              hipStream_t stream) {
  (void)in_sizes; (void)n_in; (void)out_size; (void)ws_size;
  const float* x  = (const float*)d_in[0];
  const float* Wq = (const float*)d_in[1];
  const float* bq = (const float*)d_in[2];
  const float* Wk = (const float*)d_in[3];
  const float* bk = (const float*)d_in[4];
  const float* Wv = (const float*)d_in[5];
  const float* bv = (const float*)d_in[6];
  const float* Wo = (const float*)d_in[7];
  const float* bo = (const float*)d_in[8];
  float* out = (float*)d_out;

  const size_t MS = (size_t)BATCH * S_LEN;        // 8192 rows
  char* ws = (char*)d_ws;
  size_t off = 0;
  bf16* xb  = (bf16*)(ws + off); off += MS * DIM * sizeof(bf16);
  bf16* Wqb = (bf16*)(ws + off); off += (size_t)DIM * DIM * sizeof(bf16);
  bf16* Wkb = (bf16*)(ws + off); off += (size_t)DIM * DIM * sizeof(bf16);
  bf16* Wvb = (bf16*)(ws + off); off += (size_t)DIM * DIM * sizeof(bf16);
  bf16* Wob = (bf16*)(ws + off); off += (size_t)DIM * DIM * sizeof(bf16);
  bf16* Qb  = (bf16*)(ws + off); off += MS * DIM * sizeof(bf16);
  bf16* Kb  = (bf16*)(ws + off); off += MS * DIM * sizeof(bf16);
  bf16* Vb  = (bf16*)(ws + off); off += MS * DIM * sizeof(bf16);
  bf16* AOb = (bf16*)(ws + off); off += MS * DIM * sizeof(bf16);
  bf16* Vtb = (bf16*)(ws + off); off += MS * DIM * sizeof(bf16);

  swa_cvt_bf16<<<2048, 256, 0, stream>>>(x,  xb,  (int)(MS * DIM));
  swa_cvt_bf16<<<512,  256, 0, stream>>>(Wq, Wqb, DIM * DIM);
  swa_cvt_bf16<<<512,  256, 0, stream>>>(Wk, Wkb, DIM * DIM);
  swa_cvt_bf16<<<512,  256, 0, stream>>>(Wv, Wvb, DIM * DIM);
  swa_cvt_bf16<<<512,  256, 0, stream>>>(Wo, Wob, DIM * DIM);

  dim3 gg(DIM / 128, (unsigned)(MS / 128));       // (8, 64)
  swa_gemm_bias<0><<<gg, 256, 0, stream>>>(xb, Wqb, bq, (void*)Qb);
  swa_gemm_bias<0><<<gg, 256, 0, stream>>>(xb, Wkb, bk, (void*)Kb);
  swa_gemm_bias<0><<<gg, 256, 0, stream>>>(xb, Wvb, bv, (void*)Vb);

  dim3 gt(S_LEN / 32, HD / 32, BATCH * HEADS);    // (128, 2, 32)
  swa_transpose_v<<<gt, 256, 0, stream>>>(Vb, Vtb);

  dim3 ga(WIN / 16, NBLK, BATCH * HEADS);         // (32, 8, 32)
  swa_attn<<<ga, 256, 0, stream>>>(Qb, Kb, Vtb, AOb);

  swa_gemm_bias<1><<<gg, 256, 0, stream>>>(AOb, Wob, bo, (void*)out);
}
